// S4SSM_14508399526249
// MI455X (gfx1250) — matrix operand order repaired and checked by reading, not compiled, not run on hardware
//
#include <hip/hip_runtime.h>
#include <hip/hip_bf16.h>

// ---------------- problem constants (from reference) ----------------
#define HCH   1025          // channels H = 2*512+1
#define LSEQ  2048          // sequence length
#define NST   32            // SSM states (== wave32 lanes!)
#define NLAY  4             // layers
#define KPAD  1056          // 33*32  : padded K (=H) for 16x16x32 WMMA
#define MPAD  2176          // 17*128 : padded M (=2H), whole 128-row blocks
#define NN    2048          // N (= L), 128*16
#define BM    128           // block tile M (8 waves x 16)
#define BN    128           // block tile N (8 WMMA tiles per wave)
#define BSTR  40            // LDS k-stride (32 + 8 pad) -> conflict-free b128 reads
#define KCHUNKS (KPAD / 32) // 33

typedef __attribute__((ext_vector_type(16))) __bf16 v16bf;
typedef __attribute__((ext_vector_type(8)))  __bf16 v8bf;
typedef __attribute__((ext_vector_type(8)))  float  v8f;

static __device__ __forceinline__ unsigned short f2bf(float f) {
    unsigned int u = __float_as_uint(f);
    u = (u + 0x7FFFu + ((u >> 16) & 1u)) >> 16;     // round-to-nearest-even
    return (unsigned short)u;
}

// ---------------- 0) discretize SSM params: dA = exp(dt*A), CB = C * (dA-1)/A * B
__global__ void s4_params_kernel(const float* __restrict__ log_dt,
                                 const float* __restrict__ A_log,
                                 const float* __restrict__ A_im,
                                 const float* __restrict__ B_re, const float* __restrict__ B_im,
                                 const float* __restrict__ C_re, const float* __restrict__ C_im,
                                 float* __restrict__ dAr, float* __restrict__ dAi,
                                 float* __restrict__ cbr, float* __restrict__ cbi) {
    int i = blockIdx.x * blockDim.x + threadIdx.x;
    if (i >= HCH * NST) return;
    int h = i / NST;
    float ar = -__expf(A_log[i]);
    float ai = A_im[i];
    float dt = __expf(log_dt[h]);
    float ex = __expf(dt * ar);
    float sn, cs;
    __sincosf(dt * ai, &sn, &cs);
    float dar = ex * cs, dai = ex * sn;           // dA = exp(dt*A)
    float nr = dar - 1.0f, ni = dai;              // (dA-1)
    float inv = 1.0f / (ar * ar + ai * ai);
    float qr = (nr * ar + ni * ai) * inv;         // (dA-1)/A
    float qi = (ni * ar - nr * ai) * inv;
    float br = B_re[i], bi = B_im[i];
    float dbr = qr * br - qi * bi;                // dB
    float dbi = qr * bi + qi * br;
    float crv = C_re[i], civ = C_im[i];
    cbr[i] = crv * dbr - civ * dbi;               // CB = C*dB
    cbi[i] = crv * dbi + civ * dbr;
    dAr[i] = dar;
    dAi[i] = dai;
}

// ---------------- 0b) pack Wout (2050x1025 f32) -> bf16 padded (MPAD x KPAD)
__global__ void s4_packw_kernel(const float* __restrict__ W, unsigned short* __restrict__ Wb) {
    int i = blockIdx.x * blockDim.x + threadIdx.x;
    if (i >= MPAD * KPAD) return;
    int m = i / KPAD, k = i - m * KPAD;
    float v = (m < 2 * HCH && k < HCH) ? W[m * HCH + k] : 0.0f;
    Wb[i] = f2bf(v);
}

// ---------------- 0c) zero the K-pad columns of yT (k = 1025..1055)
__global__ void s4_pady_kernel(unsigned short* __restrict__ yT) {
    int i = blockIdx.x * blockDim.x + threadIdx.x;
    const int pad = KPAD - HCH;                   // 31
    if (i >= NN * pad) return;
    int n = i / pad, k = HCH + (i - n * pad);
    yT[(unsigned)n * KPAD + k] = 0;
}

// ---------------- 1) SSM recurrence + D*u + exact GELU, wave-per-channel,
//                     lane-per-state. Writes gelu(y) transposed+bf16 (NN x KPAD).
__global__ void s4_scan_gelu_kernel(const float* __restrict__ U,
                                    const float* __restrict__ dAr, const float* __restrict__ dAi,
                                    const float* __restrict__ cbr, const float* __restrict__ cbi,
                                    const float* __restrict__ Dv,
                                    unsigned short* __restrict__ yT) {
    int wave = threadIdx.x >> 5;
    int lane = threadIdx.x & 31;
    int h = blockIdx.x * 8 + wave;
    if (h >= HCH) return;                          // wave-uniform
    int i = h * NST + lane;
    float ar = dAr[i], ai = dAi[i];
    float cr = cbr[i], ci = cbi[i];
    float d = Dv[h];
    const float* u = U + (unsigned)h * LSEQ;
    float sr = 0.0f, si = 0.0f, ybuf = 0.0f;
#pragma unroll 4
    for (int t = 0; t < LSEQ; ++t) {
        float ut  = u[t];
        float nsr = ar * sr - ai * si + ut;        // s = dA*s + u
        float nsi = ar * si + ai * sr;
        sr = nsr; si = nsi;
        float p = cr * sr - ci * si;               // Re(CB * s)
        p += __shfl_xor(p, 16, 32);                // full-wave butterfly sum
        p += __shfl_xor(p,  8, 32);
        p += __shfl_xor(p,  4, 32);
        p += __shfl_xor(p,  2, 32);
        p += __shfl_xor(p,  1, 32);
        float y = 2.0f * p + d * ut;
        y = 0.5f * y * (1.0f + erff(y * 0.70710678118654752f));   // exact GELU
        if ((t & 31) == lane) ybuf = y;            // lane-cyclic stash
        if ((t & 31) == 31)                        // 32 lanes flush together
            yT[(unsigned)(t - 31 + lane) * KPAD + h] = f2bf(ybuf);
    }
}

// ---------------- 2) G = Wout_bf16 @ yT^T via v_wmma_f32_16x16x32_bf16.
//    Block = 8 waves, 128x128 tile. Ping-pong LDS staging (one barrier per
//    chunk; next chunk's global loads issued before the 8 WMMAs). Fragments
//    loaded pairwise into fresh locals so ds_load of tile j+1 is in flight
//    while tile j's WMMA executes. Last chunk peeled (no in-loop branch).
__global__ void __launch_bounds__(256)
s4_glu_gemm_kernel(const unsigned short* __restrict__ Wb,  // MPAD x KPAD
                   const unsigned short* __restrict__ yT,  // NN x KPAD
                   float* __restrict__ G) {                // MPAD x NN
    __shared__ unsigned short Bs[2][BN * BSTR];    // 2 x 10 KB ping-pong

    int lane = threadIdx.x & 31;
    int wave = threadIdx.x >> 5;
    int mbase = blockIdx.x * BM;                   // always fully valid (MPAD = 17*128)
    int nbase = blockIdx.y * BN;

    // A fragment (16x32 bf16, from global): lane<16 -> row=lane, K {0..7,16..23};
    //                                       lane>=16 -> row=lane-16, K {8..15,24..31}
    int am = lane & 15;
    int ak = (lane >> 4) * 8;
    const unsigned short* arow = Wb + (size_t)(mbase + wave * 16 + am) * KPAD + ak;

    // B staging: thread i -> col n = i>>1, k-half = (i&1)*16 (two b128 per thread)
    int sn = threadIdx.x >> 1;
    int sk = (threadIdx.x & 1) * 16;
    const unsigned short* bsrc = yT + (size_t)(nbase + sn) * KPAD + sk;
    int bdsto = sn * BSTR + sk;                    // element offset within a buffer

    // B fragment (32x16 bf16, from LDS): lane<16 -> col=lane, K {0..7,16..23};
    //                                    lane>=16 -> col=lane-16, K {8..15,24..31}
    int bn = lane & 15;
    int bk = (lane >> 4) * 8;
    int bfrag0 = bn * BSTR + bk;                   // tile j adds j*16*BSTR

    const v8f vzero = {0.f, 0.f, 0.f, 0.f, 0.f, 0.f, 0.f, 0.f};
    v8f acc[8];
#pragma unroll
    for (int j = 0; j < 8; ++j) acc[j] = vzero;

    union V16 { v16bf v; v8bf h[2]; };

    // consume one staged 128x32 chunk: A from global, 8 WMMAs from LDS
    auto consume = [&](const unsigned short* bb, const unsigned short* arowk) {
        V16 A;
        A.h[0] = *(const v8bf*)(arowk);            // K+0..7   (16B)
        A.h[1] = *(const v8bf*)(arowk + 16);       // K+16..23 (16B)
#pragma unroll
        for (int j = 0; j < 8; j += 2) {
            const unsigned short* bp0 = bb + bfrag0 + j * 16 * BSTR;
            const unsigned short* bp1 = bp0 + 16 * BSTR;
            V16 B0, B1;
            B0.h[0] = *(const v8bf*)(bp0);         // 4x ds_load_b128 before
            B0.h[1] = *(const v8bf*)(bp0 + 16);    // the first wmma below
            B1.h[0] = *(const v8bf*)(bp1);
            B1.h[1] = *(const v8bf*)(bp1 + 16);
            acc[j]     = __builtin_amdgcn_wmma_f32_16x16x32_bf16(
                false, A.v, false, B0.v, (short)0, acc[j],     false, false);
            acc[j + 1] = __builtin_amdgcn_wmma_f32_16x16x32_bf16(
                false, A.v, false, B1.v, (short)0, acc[j + 1], false, false);
        }
    };

    // ---- prologue: stage chunk 0 into Bs[0]
    {
        v8bf g0 = *(const v8bf*)(bsrc);
        v8bf g1 = *(const v8bf*)(bsrc + 8);
        *(v8bf*)(&Bs[0][bdsto])     = g0;
        *(v8bf*)(&Bs[0][bdsto + 8]) = g1;
    }
    __syncthreads();

    // ---- steady state: 32 chunks with ping-pong staging
    int k0 = 0;
#pragma unroll 1
    for (int it = 0; it < KCHUNKS - 1; ++it, k0 += 32) {
        v8bf g0 = *(const v8bf*)(bsrc + k0 + 32);  // next chunk, issued early
        v8bf g1 = *(const v8bf*)(bsrc + k0 + 40);
        __builtin_prefetch(arow + k0 + 32, 0, 0);  // -> global_prefetch_b8

        consume(&Bs[it & 1][0], arow + k0);

        unsigned short* bd = &Bs[(it + 1) & 1][bdsto];
        *(v8bf*)(bd)     = g0;                     // park next chunk
        *(v8bf*)(bd + 8) = g1;
        __syncthreads();                           // one barrier per chunk
    }
    // ---- peeled final chunk (buf = (KCHUNKS-1)&1 = 0)
    consume(&Bs[(KCHUNKS - 1) & 1][0], arow + k0);

    // C/D layout: lane<16 -> col=lane, rows v+0..7 ; lane>=16 -> col=lane-16, rows v+8..15
    int col = lane & 15;
    int radd = (lane >> 4) * 8;
#pragma unroll
    for (int j = 0; j < 8; ++j) {
        int n = nbase + j * 16 + col;
#pragma unroll
        for (int v = 0; v < 8; ++v)
            G[(size_t)(mbase + wave * 16 + v + radd) * NN + n] = acc[j][v];
    }
}

// ---------------- 3) GLU + residual (in place): u += (g1+b1)*sigmoid(g2+b2)
__global__ void s4_glu_epilogue_kernel(float* __restrict__ U,
                                       const float* __restrict__ G,
                                       const float* __restrict__ bout) {
    int i = blockIdx.x * blockDim.x + threadIdx.x;
    if (i >= HCH * LSEQ) return;
    int h = i / LSEQ, t = i - h * LSEQ;
    float g1 = G[(size_t)h * NN + t] + bout[h];
    float g2 = G[(size_t)(h + HCH) * NN + t] + bout[h + HCH];
    float sg = 1.0f / (1.0f + __expf(-g2));
    U[i] = U[i] + g1 * sg;
}

// ---------------- host orchestration ----------------
static inline size_t align256(size_t x) { return (x + 255) & ~(size_t)255; }

extern "C" void kernel_launch(void* const* d_in, const int* in_sizes, int n_in,
                              void* d_out, int out_size, void* d_ws, size_t ws_size,
                              hipStream_t stream) {
    (void)in_sizes; (void)n_in; (void)out_size; (void)ws_size;
    const float* Z      = (const float*)d_in[0];
    const float* log_dt = (const float*)d_in[1];
    const float* A_log  = (const float*)d_in[2];
    const float* A_im   = (const float*)d_in[3];
    const float* B_re   = (const float*)d_in[4];
    const float* B_im   = (const float*)d_in[5];
    const float* C_re   = (const float*)d_in[6];
    const float* C_im   = (const float*)d_in[7];
    const float* Dv     = (const float*)d_in[8];
    const float* Wout   = (const float*)d_in[9];
    const float* bout   = (const float*)d_in[10];
    // d_in[11] = l (== 4, per reference constants)

    char* ws = (char*)d_ws;
    size_t off = 0;
    const size_t hn = (size_t)HCH * NST * sizeof(float);
    float* dAr = (float*)(ws + off); off = align256(off + hn);
    float* dAi = (float*)(ws + off); off = align256(off + hn);
    float* cbr = (float*)(ws + off); off = align256(off + hn);
    float* cbi = (float*)(ws + off); off = align256(off + hn);
    unsigned short* Wb = (unsigned short*)(ws + off);
    off = align256(off + (size_t)MPAD * KPAD * sizeof(unsigned short));
    unsigned short* yT = (unsigned short*)(ws + off);
    off = align256(off + (size_t)NN * KPAD * sizeof(unsigned short));
    float* G = (float*)(ws + off);
    off = align256(off + (size_t)MPAD * NN * sizeof(float));
    float* U = (float*)(ws + off);
    off = align256(off + (size_t)HCH * LSEQ * sizeof(float));

    // one-time (per call) preprocessing
    s4_params_kernel<<<(HCH * NST + 255) / 256, 256, 0, stream>>>(
        log_dt, A_log, A_im, B_re, B_im, C_re, C_im, dAr, dAi, cbr, cbi);
    s4_packw_kernel<<<(MPAD * KPAD + 255) / 256, 256, 0, stream>>>(Wout, Wb);
    s4_pady_kernel<<<(NN * (KPAD - HCH) + 255) / 256, 256, 0, stream>>>(yT);
    hipMemcpyAsync(U, Z, (size_t)HCH * LSEQ * sizeof(float),
                   hipMemcpyDeviceToDevice, stream);

    for (int layer = 0; layer < NLAY; ++layer) {
        s4_scan_gelu_kernel<<<dim3((HCH + 7) / 8), 256, 0, stream>>>(
            U, dAr, dAi, cbr, cbi, Dv, yT);
        s4_glu_gemm_kernel<<<dim3(MPAD / BM, NN / BN), 256, 0, stream>>>(
            Wb, yT, G);
        s4_glu_epilogue_kernel<<<(HCH * LSEQ + 255) / 256, 256, 0, stream>>>(
            U, G, bout);
    }

    hipMemcpyAsync(d_out, U, (size_t)HCH * LSEQ * sizeof(float),
                   hipMemcpyDeviceToDevice, stream);
}
